// GraphLinkGAT_68736656605835
// MI455X (gfx1250) — matrix operand order repaired and checked by reading, not compiled, not run on hardware
//
#include <hip/hip_runtime.h>
#include <hip/hip_bf16.h>
#include <math.h>

typedef __attribute__((ext_vector_type(16))) _Float16 v16h;
typedef __attribute__((ext_vector_type(8)))  float    v8f;
typedef __attribute__((ext_vector_type(4)))  float    v4f;

#define B_    32
#define N_    64
#define L_    4
#define V_    2048
#define E_    64512
#define EPG_  2016
#define M_    5

// workspace layout (float offsets)
#define OFF_H     0
#define OFF_HIN   (524288)                 // V*512 (also holds padded V*288 pre1 input)
#define OFF_XL    (OFF_HIN + 1048576)
#define OFF_XR    (OFF_XL + 524288)
#define OFF_GAT   (OFF_XR + 524288)
#define OFF_TMP   (OFF_GAT + 524288)
#define OFF_EVEC  (OFF_TMP + 524288)       // B*64*5
#define OFF_XS    (OFF_EVEC + B_*N_*M_)    // B*5*256
#define OFF_XST   (OFF_XS + B_*M_*256)
#define OFF_MISC  (OFF_XST + B_*M_*256)    // [0]=mean_ind [1]=mu [2]=rstd
#define OFF_WPK   (OFF_MISC + 16)          // packed f16 weights (32B aligned)

// packed-weight offsets (in halves) inside WPK region
#define WP_PRE1   0                        // Kp=288 -> 73728
#define WP_PRE2   73728                    // Kp=256 -> 65536
#define WP_LAYER  139264                   // per layer: Wl,Wr (512x256) + d2,d3 (256x256)
#define WP_LSZ    393216
#define WP_POST1  (WP_LAYER + 4*WP_LSZ)
#define WP_POST2  (WP_POST1 + 65536)

__device__ __forceinline__ float wave_sum32(float x) {
  #pragma unroll
  for (int off = 16; off >= 1; off >>= 1) x += __shfl_xor(x, off, 32);
  return x;
}
__device__ __forceinline__ int eidx_dev(int i, int j) { // i<j, upper-triangular edge id
  return i * (127 - i) / 2 + (j - i - 1);
}

// ---------------- mean of edge indicators ----------------
__global__ void k_mean_ind(const float* __restrict__ ind, float* __restrict__ misc) {
  __shared__ float sd[256];
  float s = 0.f;
  for (int i = threadIdx.x; i < E_; i += 256) s += ind[i];
  sd[threadIdx.x] = s; __syncthreads();
  for (int o = 128; o > 0; o >>= 1) {
    if (threadIdx.x < o) sd[threadIdx.x] += sd[threadIdx.x + o];
    __syncthreads();
  }
  if (threadIdx.x == 0) misc[0] = sd[0] / (float)E_;
}

// ---------------- weight packer: f32 KxN(256) -> f16 WMMA-B layout, K padded to mult of 32 ----
// Wp[(k>>5)*8192 + col*32 + (k&31)] = W[k][col]  (zero beyond Ksrc)
__global__ void k_pack(const float* __restrict__ W, int Ksrc, _Float16* __restrict__ Wp) {
  int idx = blockIdx.x * 256 + threadIdx.x;
  int k = idx >> 8, col = idx & 255;
  float v = (k < Ksrc) ? W[k * 256 + col] : 0.f;
  Wp[((k >> 5) * 256 + col) * 32 + (k & 31)] = (_Float16)v;
}

// ---------------- time embedding + padded [x|te|0] staging (V x 288) ----------------
__global__ void k_te(const float* __restrict__ x, const float* __restrict__ t,
                     const float* __restrict__ teW, const float* __restrict__ teb,
                     float* __restrict__ A0) {
  int v = blockIdx.x, tid = threadIdx.x;
  float a = t[v];                       // T_LIMIT == 1.0
  float hp = a * 1.57079632679f;
  float te = sinf(hp) * teW[tid] + cosf(hp) * teW[256 + tid] + a * teW[512 + tid] + teb[tid];
  te = te / (1.f + __expf(-te));        // SiLU
  A0[v * 288 + 16 + tid] = te;
  if (tid < 16) {
    A0[v * 288 + tid] = x[v * 16 + tid];
    A0[v * 288 + 272 + tid] = 0.f;      // zero pad K 272->288
  }
}

// ---------------- WMMA GEMM: out[2048x256] = act(A[2048xKp] @ W + bias (+resid)) --------------
// One wave computes a 16x64 strip (4 WMMA tiles), Kp multiple of 32, no guards.
__global__ void __launch_bounds__(256) k_gemm256(
    const float* __restrict__ A, int lda, int Kp,
    const _Float16* __restrict__ Wp,
    const float* __restrict__ bias,
    const float* resid, float* __restrict__ out, int act) {
  int wave = blockIdx.x * 8 + (threadIdx.x >> 5);   // 512 waves = 128 rowtiles x 4 strips
  int lane = threadIdx.x & 31;
  int tileM = wave >> 2, tn4 = wave & 3;
  int hi = lane >> 4, ml = lane & 15;
  int row = tileM * 16 + ml;
  int colbase = tn4 * 64 + ml;                      // tile t -> col = colbase + 16*t
  const float* arow = A + row * lda + (hi ? 16 : 0);
  v8f acc0 = {}, acc1 = {}, acc2 = {}, acc3 = {};
  int chunks = Kp >> 5;
  for (int ck = 0; ck < chunks; ++ck) {
    int kc = ck << 5;
    // A fragment: contiguous k order == floats [kc+base..+15] (base = 16*hi), matching packed B
    v4f f0 = *(const v4f*)(arow + kc);
    v4f f1 = *(const v4f*)(arow + kc + 4);
    v4f f2 = *(const v4f*)(arow + kc + 8);
    v4f f3 = *(const v4f*)(arow + kc + 12);
    v16h a;
    #pragma unroll
    for (int i = 0; i < 4; ++i) {
      a[i]      = (_Float16)f0[i];
      a[4 + i]  = (_Float16)f1[i];
      a[8 + i]  = (_Float16)f2[i];
      a[12 + i] = (_Float16)f3[i];
    }
    // B fragments: 16 contiguous halves per lane per tile (32B vector loads)
    const _Float16* bbase = Wp + (size_t)ck * 8192 + hi * 16;
    v16h b0 = *(const v16h*)(bbase + (colbase     ) * 32);
    v16h b1 = *(const v16h*)(bbase + (colbase + 16) * 32);
    v16h b2 = *(const v16h*)(bbase + (colbase + 32) * 32);
    v16h b3 = *(const v16h*)(bbase + (colbase + 48) * 32);
    acc0 = __builtin_amdgcn_wmma_f32_16x16x32_f16(false, a, false, b0, (short)0, acc0, false, false);
    acc1 = __builtin_amdgcn_wmma_f32_16x16x32_f16(false, a, false, b1, (short)0, acc1, false, false);
    acc2 = __builtin_amdgcn_wmma_f32_16x16x32_f16(false, a, false, b2, (short)0, acc2, false, false);
    acc3 = __builtin_amdgcn_wmma_f32_16x16x32_f16(false, a, false, b3, (short)0, acc3, false, false);
  }
  #pragma unroll
  for (int t = 0; t < 4; ++t) {
    v8f acc = (t == 0) ? acc0 : (t == 1) ? acc1 : (t == 2) ? acc2 : acc3;
    int col = colbase + 16 * t;
    float bc = bias ? bias[col] : 0.f;
    #pragma unroll
    for (int r = 0; r < 8; ++r) {
      int rr = tileM * 16 + r + (hi ? 8 : 0);       // C/D layout: vgpr r -> M=r (+8 hi lanes)
      float v = acc[r] + bc;
      if (resid) v += resid[rr * 256 + col];
      if (act) v = fmaxf(v, 0.f);
      out[rr * 256 + col] = v;
    }
  }
}

// ---------------- batched Jacobi eigh of normalized Laplacian; emit 5 smallest-ev columns ------
__global__ void __launch_bounds__(256) k_eigh(const float* __restrict__ ind, float* __restrict__ evecs) {
  __shared__ float A[64 * 64];
  __shared__ float Vm[64 * 64];
  __shared__ float cs_c[32], cs_s[32];
  __shared__ int top[32], bot[32], pp[32], qq[32];
  __shared__ float dvec[64];
  __shared__ int sel[5];
  int b = blockIdx.x, tid = threadIdx.x;
  for (int idx = tid; idx < 4096; idx += 256) {     // adjacency (temp in Vm)
    int i = idx >> 6, j = idx & 63;
    float v = 0.f;
    if (i != j) {
      int a0 = (i < j) ? i : j, a1 = (i < j) ? j : i;
      v = ind[b * EPG_ + eidx_dev(a0, a1)];
    }
    Vm[idx] = v;
  }
  __syncthreads();
  if (tid < 64) {
    float d = 0.f;
    for (int j = 0; j < 64; ++j) d += Vm[tid * 64 + j];
    dvec[tid] = rsqrtf(d + 1e-6f);
  }
  __syncthreads();
  for (int idx = tid; idx < 4096; idx += 256) {
    int i = idx >> 6, j = idx & 63;
    A[idx] = ((i == j) ? 1.f : 0.f) - dvec[i] * Vm[idx] * dvec[j];
  }
  __syncthreads();
  for (int idx = tid; idx < 4096; idx += 256) {
    int i = idx >> 6, j = idx & 63;
    Vm[idx] = (i == j) ? 1.f : 0.f;
  }
  if (tid < 32) { top[tid] = tid; bot[tid] = 32 + tid; }
  __syncthreads();
  for (int sweep = 0; sweep < 12; ++sweep) {
    for (int step = 0; step < 63; ++step) {
      if (tid < 32) {
        int p = top[tid], q = bot[tid];
        if (p > q) { int tt = p; p = q; q = tt; }
        pp[tid] = p; qq[tid] = q;
        float app = A[p * 64 + p], aqq = A[q * 64 + q], apq = A[p * 64 + q];
        float c = 1.f, s = 0.f;
        if (fabsf(apq) > 1e-12f) {
          float th = (aqq - app) / (2.f * apq);
          float tr = 1.f / (fabsf(th) + sqrtf(th * th + 1.f));
          if (th < 0.f) tr = -tr;
          c = rsqrtf(tr * tr + 1.f);
          s = tr * c;
        }
        cs_c[tid] = c; cs_s[tid] = s;
      }
      __syncthreads();
      for (int idx = tid; idx < 2048; idx += 256) {  // column pass (A and V): disjoint columns
        int k = idx >> 6, r = idx & 63;
        int p = pp[k], q = qq[k];
        float c = cs_c[k], s = cs_s[k];
        float arp = A[r * 64 + p], arq = A[r * 64 + q];
        A[r * 64 + p] = c * arp - s * arq;
        A[r * 64 + q] = s * arp + c * arq;
        float vrp = Vm[r * 64 + p], vrq = Vm[r * 64 + q];
        Vm[r * 64 + p] = c * vrp - s * vrq;
        Vm[r * 64 + q] = s * vrp + c * vrq;
      }
      __syncthreads();
      for (int idx = tid; idx < 2048; idx += 256) {  // row pass: disjoint rows
        int k = idx >> 6, r = idx & 63;
        int p = pp[k], q = qq[k];
        float c = cs_c[k], s = cs_s[k];
        float apr = A[p * 64 + r], aqr = A[q * 64 + r];
        A[p * 64 + r] = c * apr - s * aqr;
        A[q * 64 + r] = s * apr + c * aqr;
      }
      __syncthreads();
      if (tid == 0) {                                // round-robin rotate (keep top[0])
        int tmp = top[1];
        for (int k = 1; k < 31; ++k) top[k] = top[k + 1];
        top[31] = bot[31];
        for (int k = 31; k > 0; --k) bot[k] = bot[k - 1];
        bot[0] = tmp;
      }
      __syncthreads();
    }
  }
  if (tid < 64) dvec[tid] = A[tid * 64 + tid];
  __syncthreads();
  if (tid == 0) {
    unsigned long long used = 0ull;
    for (int m = 0; m < 5; ++m) {
      int best = 0; float bv = 1e30f;
      for (int j = 0; j < 64; ++j)
        if (!((used >> j) & 1ull) && dvec[j] < bv) { bv = dvec[j]; best = j; }
      used |= (1ull << best); sel[m] = best;
    }
  }
  __syncthreads();
  for (int idx = tid; idx < 64 * 5; idx += 256) {
    int n = idx / 5, m = idx % 5;
    evecs[b * (N_ * M_) + n * M_ + m] = Vm[n * 64 + sel[m]];
  }
}

// ---------------- spectral block ----------------
__global__ void k_spec_xs(const float* __restrict__ h, const float* __restrict__ ev,
                          float* __restrict__ xs) {
  int bm = blockIdx.x, b = bm / M_, m = bm % M_, d = threadIdx.x;
  float s = 0.f;
  for (int n = 0; n < 64; ++n)
    s += ev[b * 320 + n * 5 + m] * h[(b * 64 + n) * 256 + d];
  xs[bm * 256 + d] = s;
}
__global__ void k_spec_mm(const float* __restrict__ xs, const float* __restrict__ Wsp,
                          const float* __restrict__ bsp, float* __restrict__ xst) {
  __shared__ float row[256];
  int bm = blockIdx.x, d = threadIdx.x;
  row[d] = xs[bm * 256 + d];
  __syncthreads();
  float s = bsp[d];
  for (int k = 0; k < 256; ++k) s += row[k] * Wsp[k * 256 + d];
  xst[bm * 256 + d] = s;
}
__global__ void k_spec_hin(const float* __restrict__ h, const float* __restrict__ ev,
                           const float* __restrict__ xst, float* __restrict__ hin) {
  int v = blockIdx.x, b = v >> 6, n = v & 63, d = threadIdx.x;
  float s = 0.f;
  #pragma unroll
  for (int m = 0; m < 5; ++m)
    s += ev[b * 320 + n * 5 + m] * xst[(b * 5 + m) * 256 + d];
  hin[v * 512 + d] = h[v * 256 + d];
  hin[v * 512 + 256 + d] = s;
}

// ---------------- fused GAT edge attention: one block per destination, one wave per head -------
__global__ void k_gat(const float* __restrict__ xl, const float* __restrict__ xr,
                      const float* __restrict__ ind, const float* __restrict__ misc,
                      const float* __restrict__ We, const float* __restrict__ att,
                      float* __restrict__ gat) {
  __shared__ float xrv[256], wev[256], attv[256];
  int v = blockIdx.x, b = v >> 6, j = v & 63, tid = threadIdx.x;
  xrv[tid]  = xr[v * 256 + tid];
  wev[tid]  = We[tid];
  attv[tid] = att[tid];
  __syncthreads();
  float meanind = misc[0];
  float mx = -1e30f, ssum = 0.f, acc = 0.f;
  int base = b << 6;
  for (int i = 0; i <= j; ++i) {                  // sources i<j (edges) + self loop i==j
    float indv = (i == j) ? meanind : ind[b * EPG_ + eidx_dev(i, j)];
    float xli = xl[(base + i) * 256 + tid];
    float mval = xli + xrv[tid] + indv * wev[tid];
    mval = (mval > 0.f) ? mval : 0.2f * mval;     // leaky_relu(0.2)
    float sc = wave_sum32(mval * attv[tid]);      // per-head score (wave == head)
    if (sc > mx) {                                // online softmax
      float f = __expf(mx - sc);
      ssum = ssum * f + 1.f;
      acc  = acc  * f + xli;
      mx = sc;
    } else {
      float w = __expf(sc - mx);
      ssum += w;
      acc  += w * xli;
    }
  }
  gat[v * 256 + tid] = acc / ssum;
}

// ---------------- y = h + relu(gat + bias) (into gat) ----------------
__global__ void k_addrelu(const float* __restrict__ h, float* __restrict__ gat,
                          const float* __restrict__ gb) {
  int idx = blockIdx.x * 256 + threadIdx.x;
  int col = idx & 255;
  float g = gat[idx] + gb[col];
  gat[idx] = h[idx] + fmaxf(g, 0.f);
}

// ---------------- global-scalar layernorm: stats then apply ----------------
__global__ void __launch_bounds__(1024) k_stats(const float* __restrict__ y, float* __restrict__ misc) {
  __shared__ double s1[1024], s2[1024];
  int tid = threadIdx.x;
  double a = 0.0, c = 0.0;
  for (int i = tid; i < V_ * 256; i += 1024) { double xv = (double)y[i]; a += xv; c += xv * xv; }
  s1[tid] = a; s2[tid] = c; __syncthreads();
  for (int o = 512; o > 0; o >>= 1) {
    if (tid < o) { s1[tid] += s1[tid + o]; s2[tid] += s2[tid + o]; }
    __syncthreads();
  }
  if (tid == 0) {
    double n = (double)(V_ * 256);
    double mu = s1[0] / n;
    double var = s2[0] / n - mu * mu;
    if (var < 0.0) var = 0.0;
    misc[1] = (float)mu;
    misc[2] = (float)(1.0 / sqrt(var + 1e-5));
  }
}
__global__ void k_lnapply(const float* __restrict__ y, const float* __restrict__ misc,
                          const float* __restrict__ g, const float* __restrict__ be,
                          float* __restrict__ out) {
  int idx = blockIdx.x * 256 + threadIdx.x;
  int col = idx & 255;
  out[idx] = (y[idx] - misc[1]) * misc[2] * g[col] + be[col];
}

// ---------------- link head: one wave per edge ----------------
__global__ void k_link(const float* __restrict__ h, const int* __restrict__ fei,
                       const float* __restrict__ lw, const float* __restrict__ lb,
                       float* __restrict__ out) {
  int e = blockIdx.x * 8 + (threadIdx.x >> 5);
  int lane = threadIdx.x & 31;
  int s = fei[e], d = fei[E_ + e];
  float acc = 0.f;
  #pragma unroll
  for (int k = lane; k < 256; k += 32)
    acc += h[s * 256 + k] * h[d * 256 + k] * lw[k];
  acc = wave_sum32(acc);
  if (lane == 0) out[e] = 1.f / (1.f + __expf(-(acc + lb[0])));
}

extern "C" void kernel_launch(void* const* d_in, const int* in_sizes, int n_in,
                              void* d_out, int out_size, void* d_ws, size_t ws_size,
                              hipStream_t stream) {
  const float* x     = (const float*)d_in[0];
  const float* t     = (const float*)d_in[1];
  const int*   fei   = (const int*)  d_in[2];
  const float* ind   = (const float*)d_in[3];
  /* d_in[4] = batch (unused; b = v>>6) */
  const float* teW   = (const float*)d_in[5];
  const float* teb   = (const float*)d_in[6];
  const float* pre1W = (const float*)d_in[7];
  const float* pre1b = (const float*)d_in[8];
  const float* pre2W = (const float*)d_in[9];
  const float* pre2b = (const float*)d_in[10];
  const float* specW = (const float*)d_in[11];
  const float* specb = (const float*)d_in[12];
  const float* Wl    = (const float*)d_in[13];
  const float* bl    = (const float*)d_in[14];
  const float* Wr    = (const float*)d_in[15];
  const float* br    = (const float*)d_in[16];
  const float* We    = (const float*)d_in[17];
  const float* att   = (const float*)d_in[18];
  const float* gatb  = (const float*)d_in[19];
  const float* n1g   = (const float*)d_in[20];
  const float* n1b   = (const float*)d_in[21];
  const float* d2W   = (const float*)d_in[22];
  const float* d2b   = (const float*)d_in[23];
  const float* n2g   = (const float*)d_in[24];
  const float* n2b   = (const float*)d_in[25];
  const float* d3W   = (const float*)d_in[26];
  const float* d3b   = (const float*)d_in[27];
  const float* n3g   = (const float*)d_in[28];
  const float* n3b   = (const float*)d_in[29];
  const float* post1W= (const float*)d_in[30];
  const float* post1b= (const float*)d_in[31];
  const float* post2W= (const float*)d_in[32];
  const float* post2b= (const float*)d_in[33];
  const float* linkW = (const float*)d_in[34];
  const float* linkb = (const float*)d_in[35];

  float* ws  = (float*)d_ws;
  float* H   = ws + OFF_H;
  float* HIN = ws + OFF_HIN;
  float* XL  = ws + OFF_XL;
  float* XR  = ws + OFF_XR;
  float* GAT = ws + OFF_GAT;
  float* TMP = ws + OFF_TMP;
  float* EV  = ws + OFF_EVEC;
  float* XS  = ws + OFF_XS;
  float* XST = ws + OFF_XST;
  float* MISC= ws + OFF_MISC;
  _Float16* WPK = (_Float16*)(ws + OFF_WPK);

  // ---- pack all weights into f16 WMMA-B layout (branch-free GEMM inner loops) ----
  k_pack<<<288, 256, 0, stream>>>(pre1W, 272, WPK + WP_PRE1);
  k_pack<<<256, 256, 0, stream>>>(pre2W, 256, WPK + WP_PRE2);
  for (int l = 0; l < L_; ++l) {
    _Float16* wl = WPK + WP_LAYER + l * WP_LSZ;
    k_pack<<<512, 256, 0, stream>>>(Wl  + l * 131072, 512, wl);
    k_pack<<<512, 256, 0, stream>>>(Wr  + l * 131072, 512, wl + 131072);
    k_pack<<<256, 256, 0, stream>>>(d2W + l * 65536,  256, wl + 262144);
    k_pack<<<256, 256, 0, stream>>>(d3W + l * 65536,  256, wl + 327680);
  }
  k_pack<<<256, 256, 0, stream>>>(post1W, 256, WPK + WP_POST1);
  k_pack<<<256, 256, 0, stream>>>(post2W, 256, WPK + WP_POST2);

  k_mean_ind<<<1, 256, 0, stream>>>(ind, MISC);
  k_te<<<V_, 256, 0, stream>>>(x, t, teW, teb, HIN);
  k_gemm256<<<64, 256, 0, stream>>>(HIN, 288, 288, WPK + WP_PRE1, pre1b, nullptr, TMP, 1);
  k_gemm256<<<64, 256, 0, stream>>>(TMP, 256, 256, WPK + WP_PRE2, pre2b, nullptr, H, 1);
  k_eigh<<<B_, 256, 0, stream>>>(ind, EV);

  for (int l = 0; l < L_; ++l) {
    _Float16* wl = WPK + WP_LAYER + l * WP_LSZ;
    k_spec_xs<<<B_ * M_, 256, 0, stream>>>(H, EV, XS);
    k_spec_mm<<<B_ * M_, 256, 0, stream>>>(XS, specW + l * 65536, specb + l * 256, XST);
    k_spec_hin<<<V_, 256, 0, stream>>>(H, EV, XST, HIN);
    k_gemm256<<<64, 256, 0, stream>>>(HIN, 512, 512, wl,          bl + l * 256, nullptr, XL, 0);
    k_gemm256<<<64, 256, 0, stream>>>(HIN, 512, 512, wl + 131072, br + l * 256, nullptr, XR, 0);
    k_gat<<<V_, 256, 0, stream>>>(XL, XR, ind, MISC, We + l * 256, att + l * 256, GAT);
    k_addrelu<<<V_, 256, 0, stream>>>(H, GAT, gatb + l * 256);
    k_stats<<<1, 1024, 0, stream>>>(GAT, MISC);
    k_lnapply<<<V_, 256, 0, stream>>>(GAT, MISC, n1g + l * 256, n1b + l * 256, H);
    k_gemm256<<<64, 256, 0, stream>>>(H, 256, 256, wl + 262144, d2b + l * 256, H, TMP, 0);
    k_stats<<<1, 1024, 0, stream>>>(TMP, MISC);
    k_lnapply<<<V_, 256, 0, stream>>>(TMP, MISC, n2g + l * 256, n2b + l * 256, H);
    k_gemm256<<<64, 256, 0, stream>>>(H, 256, 256, wl + 327680, d3b + l * 256, H, TMP, 0);
    k_stats<<<1, 1024, 0, stream>>>(TMP, MISC);
    k_lnapply<<<V_, 256, 0, stream>>>(TMP, MISC, n3g + l * 256, n3b + l * 256, H);
  }

  k_gemm256<<<64, 256, 0, stream>>>(H, 256, 256, WPK + WP_POST1, post1b, nullptr, TMP, 1);
  k_gemm256<<<64, 256, 0, stream>>>(TMP, 256, 256, WPK + WP_POST2, post2b, nullptr, H, 1);
  k_link<<<E_ / 8, 256, 0, stream>>>(H, fei, linkW, linkb, (float*)d_out);
}
